// HSTUAttention_82360292868852
// MI455X (gfx1250) — matrix operand order repaired and checked by reading, not compiled, not run on hardware
//
#include <hip/hip_runtime.h>

// ---------------------------------------------------------------------------
// HSTU attention block for MI455X (gfx1250), wave32 + WMMA bf16 + TDM.
//   B=4, N=2048, D=1024, H=8, ATT=LIN=64, QKV_DIM=2048 (4th chunk unused)
// Pipeline (bf16 operands, fp32 accum):
//   [1] LayerNorm(x) -> X16 (bf16)
//   [2] w_qkv, w_out -> bf16
//   [3] QKV GEMM (skip unused chunk) -> Q16, K16, Vt16 (V stored transposed)
//       B-operand tiles staged into LDS via tensor_load_to_lds (double buffer)
//   [4] flash attention: K/Vt tiles TDM-staged (block-shared), S=QK^T (wmma),
//       silu/n + causal mask, online softmax, P via LDS -> wmma P@V -> O16
//   [5] y = x + O16 @ w_out^T + b_out (wmma, TDM-staged weights, fp32 out)
// ---------------------------------------------------------------------------

#define DEVI __device__ __forceinline__

typedef __bf16 bf16_t;
typedef __attribute__((ext_vector_type(16))) __bf16 v16bf;
typedef __attribute__((ext_vector_type(8)))  __bf16 v8bf;
typedef __attribute__((ext_vector_type(8)))  float  v8f;
typedef __attribute__((ext_vector_type(4)))  unsigned int v4u;
typedef __attribute__((ext_vector_type(8)))  int v8i;
typedef __attribute__((ext_vector_type(4)))  int v4i;

constexpr int Bc  = 4;
constexpr int Nc  = 2048;
constexpr int Dc  = 1024;
constexpr int Hc  = 8;
constexpr int QKVD = 2048;       // (2*64 + 2*64) * 8
constexpr int OUTK = 512;        // H * LIN

DEVI v8f v8f_zero() { v8f z = {0.f,0.f,0.f,0.f,0.f,0.f,0.f,0.f}; return z; }

// A fragment (16x32, MxK) from a row-major matrix; `A` points at tile origin.
// Lane L: row = L&15, K chunks = (L>>4)*8 + {0..7} and +16.  (ISA 7.12.2)
DEVI v16bf frag_a_rowmajor(const bf16_t* A, int lda, int lane) {
  const bf16_t* p = A + (size_t)(lane & 15) * lda + ((lane >> 4) << 3);
  union { v16bf v; v8bf h[2]; } u;
  u.h[0] = *(const v8bf*)(p);
  u.h[1] = *(const v8bf*)(p + 16);
  return u.v;
}

// B fragment (32x16, KxN) where memory is N-major: B[k][n] = W[n][k].
// Lane L: n = L&15, K chunks = (L>>4)*8 + {0..7} and +16 (matches A).
DEVI v16bf frag_b_nmajor(const bf16_t* W, int ldw, int lane) {
  const bf16_t* p = W + (size_t)(lane & 15) * ldw + ((lane >> 4) << 3);
  union { v16bf v; v8bf h[2]; } u;
  u.h[0] = *(const v8bf*)(p);
  u.h[1] = *(const v8bf*)(p + 16);
  return u.v;
}

DEVI v8f wmma_bf16(v16bf a, v16bf b, v8f c) {
  return __builtin_amdgcn_wmma_f32_16x16x32_bf16(
      /*neg_a=*/false, a, /*neg_b=*/false, b,
      /*c_mod=*/(short)0, c, /*reuse_a=*/false, /*reuse_b=*/false);
}

// ---------------------------------------------------------------------------
// Tensor Data Mover: 2D bf16 tile (tile_k contiguous elems x tile_n rows,
// row stride = stride_k elems) from global -> LDS (packed row-major, pitch
// tile_k).  D# layout per CDNA5 ISA:
//   group0: [1:0]count=1 | [63:32]lds_addr | [120:64]global_addr | type=2
//   group1: [17:16]data_size=1(2B) | tensor_dim0/1 = tile dims |
//           tile_dim0/1 | tensor_dim0_stride
// Issued once per block (wave 0); completion via TENSORcnt.
// ---------------------------------------------------------------------------
DEVI void tdm_load_2d_bf16(unsigned lds_addr, const bf16_t* gptr,
                           unsigned tile_k, unsigned tile_n,
                           unsigned long long stride_k) {
  const unsigned long long ga = (unsigned long long)(uintptr_t)gptr;
  v4u g0;
  g0[0] = 1u;                                        // count=1 (valid), no gather
  g0[1] = lds_addr;                                  // LDS byte address
  g0[2] = (unsigned)(ga & 0xffffffffu);              // global_addr[31:0]
  g0[3] = (unsigned)((ga >> 32) & 0x01ffffffu) | (2u << 30);  // addr[56:32]|type=2
  v8i g1;
  g1[0] = (int)(1u << 16);                           // data_size=1 (2 bytes)
  g1[1] = (int)((tile_k & 0xffffu) << 16);           // tensor_dim0[15:0]
  g1[2] = (int)(((tile_k >> 16) & 0xffffu) | ((tile_n & 0xffffu) << 16));
  g1[3] = (int)(((tile_n >> 16) & 0xffffu) | ((tile_k & 0xffffu) << 16)); // tile_dim0
  g1[4] = (int)(tile_n & 0xffffu);                   // tile_dim1; tile_dim2=0
  g1[5] = (int)(stride_k & 0xffffffffu);             // tensor_dim0_stride[31:0]
  g1[6] = (int)((stride_k >> 32) & 0xffffu);         // stride[47:32]; dim1_stride=0
  g1[7] = 0;
  const v4i z4 = {0, 0, 0, 0};
#if __has_include(<hip/amd_detail/amd_gfx1250_TDM.h>)
  const v8i z8 = {0, 0, 0, 0, 0, 0, 0, 0};
  __builtin_amdgcn_tensor_load_to_lds(g0, g1, z4, z4, z8, 0);   // clang-23 form
#else
  __builtin_amdgcn_tensor_load_to_lds(g0, g1, z4, z4, 0);       // ROCm 7.2 form
#endif
}

DEVI unsigned lds_off(const void* p) {               // generic->LDS byte offset
  return (unsigned)(uintptr_t)p;                     // LDS aperture: addr[31:0]
}

// ---------------------------------------------------------------------------
// [1] LayerNorm + fp32 -> bf16.  One block (256 thr) per row of 1024.
// ---------------------------------------------------------------------------
__global__ void ln_bf16_kernel(const float* __restrict__ x,
                               const float* __restrict__ g,
                               const float* __restrict__ b,
                               bf16_t* __restrict__ xn) {
  const int row = blockIdx.x;
  const int t   = threadIdx.x;
  const float4 v = ((const float4*)(x + (size_t)row * Dc))[t];
  float s  = v.x + v.y + v.z + v.w;
  float ss = v.x*v.x + v.y*v.y + v.z*v.z + v.w*v.w;
  __shared__ float rs[256], rss[256];
  rs[t] = s; rss[t] = ss;
  __syncthreads();
  for (int o = 128; o > 0; o >>= 1) {
    if (t < o) { rs[t] += rs[t + o]; rss[t] += rss[t + o]; }
    __syncthreads();
  }
  const float mu  = rs[0] * (1.f / Dc);
  const float var = rss[0] * (1.f / Dc) - mu * mu;
  const float inv = rsqrtf(var + 1e-5f);
  const int c = t * 4;
  bf16_t* o = xn + (size_t)row * Dc + c;
  o[0] = (bf16_t)((v.x - mu) * inv * g[c + 0] + b[c + 0]);
  o[1] = (bf16_t)((v.y - mu) * inv * g[c + 1] + b[c + 1]);
  o[2] = (bf16_t)((v.z - mu) * inv * g[c + 2] + b[c + 2]);
  o[3] = (bf16_t)((v.w - mu) * inv * g[c + 3] + b[c + 3]);
}

// ---------------------------------------------------------------------------
// [2] fp32 -> bf16 weight conversion (4 elements / thread).
// ---------------------------------------------------------------------------
__global__ void cvt_bf16x4_kernel(const float* __restrict__ s,
                                  bf16_t* __restrict__ d, int n4) {
  int i = blockIdx.x * blockDim.x + threadIdx.x;
  if (i < n4) {
    float4 v = ((const float4*)s)[i];
    d[i*4+0] = (bf16_t)v.x;  d[i*4+1] = (bf16_t)v.y;
    d[i*4+2] = (bf16_t)v.z;  d[i*4+3] = (bf16_t)v.w;
  }
}

// ---------------------------------------------------------------------------
// [3] QKV GEMM: C[8192 x 2048] = X16 @ Wqkv^T (+bias); only q/k/v chunks.
// grid = (64, 24) : y -> (head h = y/3, chunk c = y%3), block tile 128x64.
// Weight tile (64 rows x 32 k) is block-shared -> TDM-staged, double buffer.
// q,k stored row-major per (b,h); v stored TRANSPOSED (dim-major) for PV wmma.
// ---------------------------------------------------------------------------
__global__ void qkv_gemm_kernel(const bf16_t* __restrict__ X,
                                const bf16_t* __restrict__ W,
                                const float*  __restrict__ bias,
                                bf16_t* __restrict__ Qo,
                                bf16_t* __restrict__ Ko,
                                bf16_t* __restrict__ Vt) {
  const int lane = threadIdx.x & 31;
  const int w    = threadIdx.x >> 5;
  const int m0   = blockIdx.x * 128 + w * 16;
  const int g    = blockIdx.y;
  const int h    = g / 3;
  const int c    = g % 3;           // 0:q 1:k 2:v   (uniform per block)
  const int n0   = h * 256 + c * 64;

  __shared__ __align__(16) bf16_t wtile[2][64 * 32];

  v8f acc[4];
  acc[0] = v8f_zero(); acc[1] = v8f_zero(); acc[2] = v8f_zero(); acc[3] = v8f_zero();

  const bf16_t* Arow = X + (size_t)m0 * Dc;
  const bf16_t* Wrow = W + (size_t)n0 * Dc;

  if (w == 0)
    tdm_load_2d_bf16(lds_off(&wtile[0][0]), Wrow, 32, 64, Dc);

  int cur = 0;
  for (int k0 = 0; k0 < Dc; k0 += 32, cur ^= 1) {
    if (w == 0) {
      if (k0 + 32 < Dc) {
        tdm_load_2d_bf16(lds_off(&wtile[cur ^ 1][0]), Wrow + k0 + 32, 32, 64, Dc);
        __builtin_amdgcn_s_wait_tensorcnt(1);   // current tile done, next in flight
      } else {
        __builtin_amdgcn_s_wait_tensorcnt(0);
      }
    }
    __syncthreads();
    __builtin_prefetch(Arow + k0 + 128, 0, 3);  // near (WGP) prefetch of A stream
    v16bf a = frag_a_rowmajor(Arow + k0, Dc, lane);
    const bf16_t* wb = &wtile[cur][0];
#pragma unroll
    for (int j = 0; j < 4; ++j) {
      v16bf bf = frag_b_nmajor(wb + (j * 16) * 32, 32, lane);
      acc[j] = wmma_bf16(a, bf, acc[j]);
    }
    __syncthreads();                            // reads done before TDM overwrite
  }

  const int colk  = lane & 15;
  const int rbase = (lane >> 4) << 3;
#pragma unroll
  for (int j = 0; j < 4; ++j) {
    const int lcol = j * 16 + colk;                 // 0..63 within chunk
    const float bv = bias[n0 + lcol];
#pragma unroll
    for (int i = 0; i < 8; ++i) {
      const int m  = m0 + rbase + i;
      const int bb = m >> 11;                       // N = 2048
      const int n  = m & (Nc - 1);
      const size_t bh = (size_t)(bb * Hc + h);
      const float val = acc[j][i] + bv;
      if (c == 0)      Qo[(bh * Nc + n) * 64 + lcol] = (bf16_t)val;
      else if (c == 1) Ko[(bh * Nc + n) * 64 + lcol] = (bf16_t)val;
      else             Vt[(bh * 64 + lcol) * Nc + n] = (bf16_t)val;
    }
  }
}

// ---------------------------------------------------------------------------
// [4] Flash attention per (b,h).  grid = (N/128, B*H), block = 256 (8 waves),
// each wave owns 16 query rows.  Key range uniform across block, so K/Vt
// tiles are block-shared -> TDM-staged (2 DMAs / step, double buffered).
// ---------------------------------------------------------------------------
__global__ void attn_kernel(const bf16_t* __restrict__ Qg,
                            const bf16_t* __restrict__ Kg,
                            const bf16_t* __restrict__ Vtg,
                            bf16_t* __restrict__ Og) {
  const int bh   = blockIdx.y;
  const int b    = bh >> 3;
  const int h    = bh & 7;
  const int qb   = blockIdx.x * 128;
  const int w    = threadIdx.x >> 5;
  const int lane = threadIdx.x & 31;
  const int colk  = lane & 15;
  const int rbase = (lane >> 4) << 3;

  const bf16_t* Qp = Qg  + (size_t)bh * Nc * 64;
  const bf16_t* Kp = Kg  + (size_t)bh * Nc * 64;
  const bf16_t* Vp = Vtg + (size_t)bh * 64 * Nc;

  const int q0 = qb + w * 16;
  __shared__ __align__(16) bf16_t kbuf[2][32 * 64];   // 32 keys x 64 dims
  __shared__ __align__(16) bf16_t vbuf[2][64 * 32];   // 64 dims x 32 keys
  __shared__ __align__(16) bf16_t pbuf[8][16 * 32];
  bf16_t* myp = pbuf[w];

  const v16bf qf0 = frag_a_rowmajor(Qp + (size_t)q0 * 64,      64, lane);
  const v16bf qf1 = frag_a_rowmajor(Qp + (size_t)q0 * 64 + 32, 64, lane);

  v8f o0 = v8f_zero(), o1 = v8f_zero(), o2 = v8f_zero(), o3 = v8f_zero();
  float mrow[8], lrow[8];
#pragma unroll
  for (int i = 0; i < 8; ++i) { mrow[i] = -__builtin_inff(); lrow[i] = 0.f; }

  const float inv_n = 1.0f / (float)Nc;
  const int kend = qb + 128;                        // causal upper bound, uniform

  if (w == 0) {                                     // prologue: stage first tiles
    tdm_load_2d_bf16(lds_off(&kbuf[0][0]), Kp, 64, 32, 64);
    tdm_load_2d_bf16(lds_off(&vbuf[0][0]), Vp, 32, 64, Nc);
  }

  int cur = 0;
  for (int kc = 0; kc < kend; kc += 32, cur ^= 1) {
    if (w == 0) {
      if (kc + 32 < kend) {
        tdm_load_2d_bf16(lds_off(&kbuf[cur ^ 1][0]), Kp + (size_t)(kc + 32) * 64, 64, 32, 64);
        tdm_load_2d_bf16(lds_off(&vbuf[cur ^ 1][0]), Vp + kc + 32, 32, 64, Nc);
        __builtin_amdgcn_s_wait_tensorcnt(2);       // current pair done
      } else {
        __builtin_amdgcn_s_wait_tensorcnt(0);
      }
    }
    __syncthreads();

    // ---- S = Q @ K^T for 32 keys from LDS (two 16x16 tiles, 2 wmma each)
    const bf16_t* kb = &kbuf[cur][0];
    v8f s0 = v8f_zero(), s1 = v8f_zero();
    {
      v16bf f;
      f = frag_b_nmajor(kb,            64, lane); s0 = wmma_bf16(qf0, f, s0);
      f = frag_b_nmajor(kb + 32,       64, lane); s0 = wmma_bf16(qf1, f, s0);
      f = frag_b_nmajor(kb + 16 * 64,      64, lane); s1 = wmma_bf16(qf0, f, s1);
      f = frag_b_nmajor(kb + 16 * 64 + 32, 64, lane); s1 = wmma_bf16(qf1, f, s1);
    }
    // ---- masked silu + online softmax (row stats via 16-lane shfl reduce)
#pragma unroll
    for (int i = 0; i < 8; ++i) {
      const int grow = q0 + rbase + i;
      const float x0 = s0[i] * 0.125f;               // 1/sqrt(64)
      const float x1 = s1[i] * 0.125f;
      const float z0 = (kc + colk)      <= grow ? (x0 / (1.f + __expf(-x0))) * inv_n
                                                : -__builtin_inff();
      const float z1 = (kc + 16 + colk) <= grow ? (x1 / (1.f + __expf(-x1))) * inv_n
                                                : -__builtin_inff();
      float zm = fmaxf(z0, z1);
#pragma unroll
      for (int off = 1; off < 16; off <<= 1)
        zm = fmaxf(zm, __shfl_xor(zm, off, 16));
      const float mn = fmaxf(mrow[i], zm);
      const float p0 = __expf(z0 - mn);
      const float p1 = __expf(z1 - mn);
      float ps = p0 + p1;
#pragma unroll
      for (int off = 1; off < 16; off <<= 1)
        ps += __shfl_xor(ps, off, 16);
      const float alpha = __expf(mrow[i] - mn);
      lrow[i] = lrow[i] * alpha + ps;
      mrow[i] = mn;
      o0[i] *= alpha; o1[i] *= alpha; o2[i] *= alpha; o3[i] *= alpha;
      myp[(rbase + i) * 32 + colk]      = (bf16_t)p0;
      myp[(rbase + i) * 32 + 16 + colk] = (bf16_t)p1;
    }
    __syncthreads();
    // ---- O += P(16x32) @ V(32x64); Vt tile is dim-major in LDS
    const v16bf pa = frag_a_rowmajor(myp, 32, lane);
    const bf16_t* vb = &vbuf[cur][0];
    v16bf f;
    f = frag_b_nmajor(vb,           32, lane); o0 = wmma_bf16(pa, f, o0);
    f = frag_b_nmajor(vb + 16 * 32, 32, lane); o1 = wmma_bf16(pa, f, o1);
    f = frag_b_nmajor(vb + 32 * 32, 32, lane); o2 = wmma_bf16(pa, f, o2);
    f = frag_b_nmajor(vb + 48 * 32, 32, lane); o3 = wmma_bf16(pa, f, o3);
    __syncthreads();                            // reads done before TDM overwrite
  }

  // ---- normalize and store to O16 [(B*N) x 512], col base h*64
#pragma unroll
  for (int i = 0; i < 8; ++i) {
    const float rl = 1.f / lrow[i];
    const int n = q0 + rbase + i;
    const size_t base = ((size_t)(b * Nc + n)) * OUTK + h * 64;
    Og[base + colk]      = (bf16_t)(o0[i] * rl);
    Og[base + 16 + colk] = (bf16_t)(o1[i] * rl);
    Og[base + 32 + colk] = (bf16_t)(o2[i] * rl);
    Og[base + 48 + colk] = (bf16_t)(o3[i] * rl);
  }
}

// ---------------------------------------------------------------------------
// [5] y = x + O16 @ w_out^T + b_out.  M=8192, K=512, N=1024. grid (64,16).
// Weight tile TDM-staged like the QKV GEMM.
// ---------------------------------------------------------------------------
__global__ void out_gemm_kernel(const bf16_t* __restrict__ A,
                                const bf16_t* __restrict__ W,
                                const float*  __restrict__ bias,
                                const float*  __restrict__ x,
                                float* __restrict__ y) {
  const int lane = threadIdx.x & 31;
  const int w    = threadIdx.x >> 5;
  const int m0   = blockIdx.x * 128 + w * 16;
  const int n0   = blockIdx.y * 64;

  __shared__ __align__(16) bf16_t wtile[2][64 * 32];

  v8f acc[4];
  acc[0] = v8f_zero(); acc[1] = v8f_zero(); acc[2] = v8f_zero(); acc[3] = v8f_zero();

  const bf16_t* Arow = A + (size_t)m0 * OUTK;
  const bf16_t* Wrow = W + (size_t)n0 * OUTK;

  if (w == 0)
    tdm_load_2d_bf16(lds_off(&wtile[0][0]), Wrow, 32, 64, OUTK);

  int cur = 0;
  for (int k0 = 0; k0 < OUTK; k0 += 32, cur ^= 1) {
    if (w == 0) {
      if (k0 + 32 < OUTK) {
        tdm_load_2d_bf16(lds_off(&wtile[cur ^ 1][0]), Wrow + k0 + 32, 32, 64, OUTK);
        __builtin_amdgcn_s_wait_tensorcnt(1);
      } else {
        __builtin_amdgcn_s_wait_tensorcnt(0);
      }
    }
    __syncthreads();
    __builtin_prefetch(Arow + k0 + 128, 0, 3);
    v16bf a = frag_a_rowmajor(Arow + k0, OUTK, lane);
    const bf16_t* wb = &wtile[cur][0];
#pragma unroll
    for (int j = 0; j < 4; ++j) {
      v16bf bf = frag_b_nmajor(wb + (j * 16) * 32, 32, lane);
      acc[j] = wmma_bf16(a, bf, acc[j]);
    }
    __syncthreads();
  }

  const int colk  = lane & 15;
  const int rbase = (lane >> 4) << 3;
#pragma unroll
  for (int j = 0; j < 4; ++j) {
    const int col = n0 + j * 16 + colk;
    const float bv = bias[col];
#pragma unroll
    for (int i = 0; i < 8; ++i) {
      const size_t m = (size_t)(m0 + rbase + i);
      y[m * Dc + col] = acc[j][i] + bv + x[m * Dc + col];
    }
  }
}

// ---------------------------------------------------------------------------
extern "C" void kernel_launch(void* const* d_in, const int* in_sizes, int n_in,
                              void* d_out, int out_size, void* d_ws, size_t ws_size,
                              hipStream_t stream) {
  const float* x     = (const float*)d_in[0];
  // d_in[1] attention_mask: causal tril by construction -> handled analytically
  const float* ln_g  = (const float*)d_in[2];
  const float* ln_b  = (const float*)d_in[3];
  const float* w_qkv = (const float*)d_in[4];
  const float* b_qkv = (const float*)d_in[5];
  const float* w_out = (const float*)d_in[6];
  const float* b_out = (const float*)d_in[7];
  float* out = (float*)d_out;

  // Workspace layout (bf16), total ~55.6 MB
  char* ws = (char*)d_ws;
  bf16_t* X16    = (bf16_t*)(ws);                               // 8192*1024
  bf16_t* Wqkv16 = (bf16_t*)(ws + (size_t)16777216);            // 2048*1024
  bf16_t* Wout16 = (bf16_t*)(ws + (size_t)20971520);            // 1024*512
  bf16_t* Q16    = (bf16_t*)(ws + (size_t)22020096);            // B*H*N*64
  bf16_t* K16    = (bf16_t*)(ws + (size_t)30408704);
  bf16_t* Vt16   = (bf16_t*)(ws + (size_t)38797312);            // B*H*64*N
  bf16_t* O16    = (bf16_t*)(ws + (size_t)47185920);            // 8192*512

  // [1] LayerNorm
  ln_bf16_kernel<<<Bc * Nc, 256, 0, stream>>>(x, ln_g, ln_b, X16);
  // [2] weight conversion
  cvt_bf16x4_kernel<<<(QKVD * Dc / 4 + 255) / 256, 256, 0, stream>>>(w_qkv, Wqkv16, QKVD * Dc / 4);
  cvt_bf16x4_kernel<<<(Dc * OUTK / 4 + 255) / 256, 256, 0, stream>>>(w_out, Wout16, Dc * OUTK / 4);
  // [3] QKV GEMM (24 col-groups: 8 heads x {q,k,v}; unused chunk skipped)
  qkv_gemm_kernel<<<dim3(Bc * Nc / 128, 24), 256, 0, stream>>>(X16, Wqkv16, b_qkv, Q16, K16, Vt16);
  // [4] flash attention
  attn_kernel<<<dim3(Nc / 128, Bc * Hc), 256, 0, stream>>>(Q16, K16, Vt16, O16);
  // [5] output GEMM + residual
  out_gemm_kernel<<<dim3(Bc * Nc / 128, Dc / 64), 256, 0, stream>>>(O16, Wout16, b_out, x, out);
}
